// TriangleAttention_59176059405030
// MI455X (gfx1250) — matrix operand order repaired and checked by reading, not compiled, not run on hardware
//
#include <hip/hip_runtime.h>
#include <stdint.h>

// ---------------------------------------------------------------------------
// CDNA5 (gfx1250) fused TriangleAttention. All contractions via
// v_wmma_f32_16x16x32_bf16 (wave32). B operands stored transposed [n][k] in
// LDS so both A and B fragments load as two ds_load_b128 each. Tile staging
// uses GLOBAL_LOAD_ASYNC_TO_LDS_B128 (ASYNCcnt).
// ---------------------------------------------------------------------------

typedef __attribute__((ext_vector_type(16))) __bf16 v16bf;
typedef __attribute__((ext_vector_type(8)))  float  v8f;

#define CZ   128
#define MTOK 256
#define NH   4
#define HD   32

__device__ __forceinline__ unsigned short f2bfu(float f) {
    return __builtin_bit_cast(unsigned short, static_cast<__bf16>(f));
}
__device__ __forceinline__ float bfu2f(unsigned short s) {
    unsigned u = ((unsigned)s) << 16;
    return __builtin_bit_cast(float, u);
}

union AFrag { v16bf v; unsigned short u[16]; };
union CFrag { v8f  v; float f[8]; };

__device__ __forceinline__ v8f wmma_bf16(v16bf a, v16bf b, v8f c) {
    return __builtin_amdgcn_wmma_f32_16x16x32_bf16(
        false, a, false, b, (short)0, c, false, false);
}

// A-fragment element -> K offset within 32 (contiguous: k = 16*(lane/16) + e)
__device__ __forceinline__ int a_kidx(int lane, int e) {
    return ((lane >> 4) * 16) + e;
}

// Async global->LDS 128-bit copy (GVS form: SGPR base + 32-bit VGPR offset).
__device__ __forceinline__ void async_g2l_b128(void* ldsdst, const void* sbase,
                                               unsigned byteoff) {
    unsigned lds = (unsigned)(uintptr_t)ldsdst;   // low 32 bits = LDS offset
    __asm__ volatile("global_load_async_to_lds_b128 %0, %1, %2 offset:0"
                     :: "v"(lds), "v"(byteoff), "s"(sbase) : "memory");
}
__device__ __forceinline__ void async_wait0() {
    __asm__ volatile("s_wait_asynccnt 0x0" ::: "memory");
}

// ===========================================================================
// Kernel 1: LayerNorm + projections.
//   q_ws [n][h][m][d] (pre-scaled by D^-0.5), k_ws [n][h][m][d],
//   v_ws [n][h][d][m] (transposed), g_ws [t][c] = sigmoid(zn@Wg),
//   bias_ws [t][h] = zn@Wb (f32).
// ===========================================================================
__global__ __launch_bounds__(256) void k1_ln_proj(
    const float* __restrict__ z,   const float* __restrict__ ln_w,
    const float* __restrict__ ln_b,
    const float* __restrict__ Wq,  const float* __restrict__ Wk,
    const float* __restrict__ Wv,  const float* __restrict__ Wb,
    const float* __restrict__ Wg,
    unsigned short* __restrict__ q_ws, unsigned short* __restrict__ k_ws,
    unsigned short* __restrict__ v_ws, unsigned short* __restrict__ g_ws,
    float* __restrict__ bias_ws)
{
    __shared__ unsigned short znb[128][136];    // A operand (bf16), padded
    __shared__ unsigned short wbufT[64][136];   // W^T half: [n][k], padded

    const int tid  = threadIdx.x;
    const int lane = tid & 31;
    const int wv   = tid >> 5;          // 8 waves
    const int row0 = blockIdx.x * 128;

    // ---- LayerNorm: one wave per row, 4 elems per lane -------------------
    for (int it = 0; it < 16; ++it) {
        int r = it * 8 + wv;
        const float4 x  = ((const float4*)(z + (size_t)(row0 + r) * CZ))[lane];
        float s = x.x + x.y + x.z + x.w;
        #pragma unroll
        for (int m = 16; m >= 1; m >>= 1) s += __shfl_xor(s, m, 32);
        const float mu = s * (1.0f / CZ);
        const float dx = x.x - mu, dy = x.y - mu, dz = x.z - mu, dw = x.w - mu;
        float v2 = dx * dx + dy * dy + dz * dz + dw * dw;
        #pragma unroll
        for (int m = 16; m >= 1; m >>= 1) v2 += __shfl_xor(v2, m, 32);
        const float inv = rsqrtf(v2 * (1.0f / CZ) + 1e-5f);
        const float4 w4 = ((const float4*)ln_w)[lane];
        const float4 b4 = ((const float4*)ln_b)[lane];
        znb[r][lane * 4 + 0] = f2bfu(dx * inv * w4.x + b4.x);
        znb[r][lane * 4 + 1] = f2bfu(dy * inv * w4.y + b4.y);
        znb[r][lane * 4 + 2] = f2bfu(dz * inv * w4.z + b4.z);
        znb[r][lane * 4 + 3] = f2bfu(dw * inv * w4.w + b4.w);
    }
    __syncthreads();

    // ---- bias = zn @ Wb (128x4) -----------------------------------------
    #pragma unroll
    for (int rep = 0; rep < 2; ++rep) {
        int o = tid + rep * 256;
        int r = o >> 2, h = o & 3;
        float acc = 0.f;
        for (int k = 0; k < CZ; ++k) acc += bfu2f(znb[r][k]) * Wb[k * 4 + h];
        bias_ws[(size_t)(row0 + r) * NH + h] = acc;
    }

    // ---- four 128x128 projections through WMMA ---------------------------
    const float* Ws[4] = {Wq, Wk, Wv, Wg};
    for (int wsel = 0; wsel < 4; ++wsel) {
        const float* W = Ws[wsel];
        unsigned short* dst;
        int sh, sm, sd;
        if      (wsel == 0) { dst = q_ws; sh = 8192; sm = 32;  sd = 1;   }
        else if (wsel == 1) { dst = k_ws; sh = 8192; sm = 32;  sd = 1;   }
        else if (wsel == 2) { dst = v_ws; sh = 8192; sm = 1;   sd = 256; }
        else                { dst = g_ws; sh = 32;   sm = 128; sd = 1;   }

        for (int nhf = 0; nhf < 2; ++nhf) {
            const int n0 = nhf * 64;
            __syncthreads();                      // protect wbufT reuse
            // transpose-load 64 cols x 128 k, two k's packed per dword store
            for (int it = 0; it < 16; ++it) {
                int p  = tid + it * 256;          // 4096 (n, kpair) pairs
                int nn = p & 63, kp = p >> 6;     // kp in 0..63
                unsigned lo = f2bfu(W[(2 * kp    ) * CZ + n0 + nn]);
                unsigned hi = f2bfu(W[(2 * kp + 1) * CZ + n0 + nn]);
                *(unsigned*)&wbufT[nn][2 * kp] = lo | (hi << 16);
            }
            __syncthreads();

            const int tn = (wv & 3) * 16;
            #pragma unroll
            for (int ttl = 0; ttl < 4; ++ttl) {
                const int tm = ((wv >> 2) * 4 + ttl) * 16;
                CFrag c;
                #pragma unroll
                for (int r = 0; r < 8; ++r) c.f[r] = 0.f;

                #pragma unroll
                for (int kk = 0; kk < 4; ++kk) {
                    AFrag a, b;
                    const int ar = tm + (lane & 15);
                    #pragma unroll
                    for (int e = 0; e < 16; ++e)
                        a.u[e] = znb[ar][kk * 32 + a_kidx(lane, e)];
                    const int br = tn + (lane & 15);
                    const int kb = kk * 32 + (lane >> 4) * 16;
                    #pragma unroll
                    for (int e = 0; e < 16; ++e) b.u[e] = wbufT[br][kb + e];
                    c.v = wmma_bf16(a.v, b.v, c.v);
                }

                const int col = n0 + tn + (lane & 15);
                const int h = col >> 5, d = col & 31;
                float vals[8];
                #pragma unroll
                for (int r = 0; r < 8; ++r) vals[r] = c.f[r];
                if (wsel == 0) {                  // fold attention scale into Q
                    #pragma unroll
                    for (int r = 0; r < 8; ++r)
                        vals[r] *= 0.17677669529663687f;
                } else if (wsel == 3) {           // sigmoid via fast rcp
                    #pragma unroll
                    for (int r = 0; r < 8; ++r)
                        vals[r] = __builtin_amdgcn_rcpf(1.0f + __expf(-vals[r]));
                }
                #pragma unroll
                for (int r = 0; r < 8; ++r) {
                    const int rr = tm + r + 8 * (lane >> 4);
                    const int t  = row0 + rr;
                    const int n  = t >> 8, mcol = t & 255;
                    dst[(size_t)n * 32768 + h * sh + mcol * sm + d * sd] =
                        f2bfu(vals[r]);
                }
            }
        }
    }
}

// ===========================================================================
// Kernel 2: per-(n,h) attention.  128 threads = 4 waves; each wave owns a
// 16-row j-tile, 4 passes to cover j=0..255.  Fused gating.
// ===========================================================================
__global__ __launch_bounds__(128) void k2_attn(
    const unsigned short* __restrict__ q_ws, const unsigned short* __restrict__ k_ws,
    const unsigned short* __restrict__ v_ws, const unsigned short* __restrict__ g_ws,
    const float* __restrict__ bias_ws, const float* __restrict__ pair_mask,
    unsigned short* __restrict__ ao_ws)
{
    __shared__ unsigned short qb [MTOK][40];   // Q   [j][d]  (A operand)
    __shared__ unsigned short kbf[MTOK][40];   // K   [k][d]  (B^T operand)
    __shared__ unsigned short vtb[HD][264];    // V^T [d][k]  (B^T operand)
    __shared__ unsigned short pb[4][16][40];   // per-wave P staging
    __shared__ float biasAdj[MTOK];            // mask>0 ? bias : -1e9

    const int tid  = threadIdx.x;
    const int lane = tid & 31;
    const int wv   = tid >> 5;                 // 4 waves
    const int nb   = blockIdx.x >> 2;
    const int h    = blockIdx.x & 3;

    const size_t base = (size_t)blockIdx.x * (MTOK * HD);
    const unsigned short* qsrc = q_ws + base;
    const unsigned short* ksrc = k_ws + base;
    const unsigned short* vsrc = v_ws + base;

    // ---- async global -> LDS staging (ASYNCcnt) --------------------------
    #pragma unroll
    for (int c = tid; c < 1024; c += 128) {
        int row = c >> 2, off = (c & 3) * 8;       // [m][d] tiles
        int vrow = c >> 5, voff = (c & 31) * 8;    // [d][m] tile
        unsigned go = (unsigned)c * 16u;
        async_g2l_b128(&qb [row][off],  qsrc, go);
        async_g2l_b128(&kbf[row][off],  ksrc, go);
        async_g2l_b128(&vtb[vrow][voff], vsrc, go);
    }
    #pragma unroll
    for (int rep = 0; rep < 2; ++rep) {
        int k = tid + rep * 128;
        float bs = bias_ws[((size_t)nb * MTOK + k) * NH + h];
        float mk = pair_mask[nb * MTOK + k];
        biasAdj[k] = (mk > 0.f) ? bs : -1.0e9f;
    }
    async_wait0();
    __syncthreads();

    for (int jt = 0; jt < 4; ++jt) {
        const int jbase = (jt * 4 + wv) * 16;

        AFrag qa;
        const int ar = jbase + (lane & 15);
        #pragma unroll
        for (int e = 0; e < 16; ++e) qa.u[e] = qb[ar][a_kidx(lane, e)];

        // ---- S = (Q*scale) K^T + biasAdj ---------------------------------
        CFrag S[16];
        #pragma unroll
        for (int kt = 0; kt < 16; ++kt) {
            AFrag bfr;
            const int bc = kt * 16 + (lane & 15);       // k-token (B "n")
            const int kb = (lane >> 4) * 16;            // d (contraction)
            #pragma unroll
            for (int e = 0; e < 16; ++e) bfr.u[e] = kbf[bc][kb + e];
            #pragma unroll
            for (int r = 0; r < 8; ++r) S[kt].f[r] = 0.f;
            S[kt].v = wmma_bf16(qa.v, bfr.v, S[kt].v);
            const float ba = biasAdj[bc];
            #pragma unroll
            for (int r = 0; r < 8; ++r) S[kt].f[r] += ba;
        }

        // ---- softmax over k ----------------------------------------------
        float rinv[8];
        #pragma unroll
        for (int r = 0; r < 8; ++r) {
            float m = -3.4e38f;
            #pragma unroll
            for (int kt = 0; kt < 16; ++kt) m = fmaxf(m, S[kt].f[r]);
            #pragma unroll
            for (int x = 8; x >= 1; x >>= 1) m = fmaxf(m, __shfl_xor(m, x, 32));
            float s = 0.f;
            #pragma unroll
            for (int kt = 0; kt < 16; ++kt) {
                float e = __expf(S[kt].f[r] - m);
                S[kt].f[r] = e;
                s += e;
            }
            #pragma unroll
            for (int x = 8; x >= 1; x >>= 1) s += __shfl_xor(s, x, 32);
            rinv[r] = __builtin_amdgcn_rcpf(s);
        }

        // ---- O = P @ V ----------------------------------------------------
        CFrag O[2];
        #pragma unroll
        for (int dn = 0; dn < 2; ++dn)
            #pragma unroll
            for (int r = 0; r < 8; ++r) O[dn].f[r] = 0.f;

        for (int kt2 = 0; kt2 < 8; ++kt2) {
            #pragma unroll
            for (int half = 0; half < 2; ++half) {
                const int kt  = kt2 * 2 + half;
                const int col = half * 16 + (lane & 15);
                #pragma unroll
                for (int r = 0; r < 8; ++r) {
                    const int row = r + 8 * (lane >> 4);
                    pb[wv][row][col] = f2bfu(S[kt].f[r] * rinv[r]);
                }
            }
            __asm__ volatile("s_wait_dscnt 0" ::: "memory");

            AFrag pa;
            const int pr = lane & 15;
            #pragma unroll
            for (int e = 0; e < 16; ++e) pa.u[e] = pb[wv][pr][a_kidx(lane, e)];

            #pragma unroll
            for (int dn = 0; dn < 2; ++dn) {
                AFrag bv;
                const int d  = dn * 16 + (lane & 15);
                const int kb = kt2 * 32 + (lane >> 4) * 16;
                #pragma unroll
                for (int e = 0; e < 16; ++e) bv.u[e] = vtb[d][kb + e];
                O[dn].v = wmma_bf16(pa.v, bv.v, O[dn].v);
            }
        }

        // ---- fused gate + store ------------------------------------------
        #pragma unroll
        for (int dn = 0; dn < 2; ++dn) {
            const int d = dn * 16 + (lane & 15);
            const int c = h * HD + d;
            #pragma unroll
            for (int r = 0; r < 8; ++r) {
                const int j = jbase + r + 8 * (lane >> 4);
                const size_t t = (size_t)nb * MTOK + j;
                const float g = bfu2f(g_ws[t * CZ + c]);
                ao_ws[t * CZ + c] = f2bfu(O[dn].f[r] * g);
            }
        }
    }
}

// ===========================================================================
// Kernel 3: out = (gated attn) @ Wo + bo  (f32), Wo staged in 64-col halves.
// ===========================================================================
__global__ __launch_bounds__(256) void k3_outproj(
    const unsigned short* __restrict__ ao_ws, const float* __restrict__ Wo,
    const float* __restrict__ bo, float* __restrict__ out)
{
    __shared__ unsigned short abuf[128][136];
    __shared__ unsigned short wbufT[64][136];

    const int tid  = threadIdx.x;
    const int lane = tid & 31;
    const int wv   = tid >> 5;
    const size_t row0 = (size_t)blockIdx.x * 128;

    const unsigned short* asrc = ao_ws + row0 * CZ;
    #pragma unroll
    for (int c = tid; c < 2048; c += 256) {
        int row = c >> 4, off = (c & 15) * 8;
        async_g2l_b128(&abuf[row][off], asrc, (unsigned)c * 16u);
    }
    async_wait0();

    for (int nhf = 0; nhf < 2; ++nhf) {
        const int n0 = nhf * 64;
        __syncthreads();
        for (int it = 0; it < 16; ++it) {
            int p  = tid + it * 256;
            int nn = p & 63, kp = p >> 6;
            unsigned lo = f2bfu(Wo[(2 * kp    ) * CZ + n0 + nn]);
            unsigned hi = f2bfu(Wo[(2 * kp + 1) * CZ + n0 + nn]);
            *(unsigned*)&wbufT[nn][2 * kp] = lo | (hi << 16);
        }
        __syncthreads();

        const int tn = (wv & 3) * 16;
        #pragma unroll
        for (int ttl = 0; ttl < 4; ++ttl) {
            const int tm = ((wv >> 2) * 4 + ttl) * 16;
            CFrag c;
            #pragma unroll
            for (int r = 0; r < 8; ++r) c.f[r] = 0.f;
            #pragma unroll
            for (int kk = 0; kk < 4; ++kk) {
                AFrag a, b;
                const int ar = tm + (lane & 15);
                #pragma unroll
                for (int e = 0; e < 16; ++e)
                    a.u[e] = abuf[ar][kk * 32 + a_kidx(lane, e)];
                const int br = tn + (lane & 15);
                const int kb = kk * 32 + (lane >> 4) * 16;
                #pragma unroll
                for (int e = 0; e < 16; ++e) b.u[e] = wbufT[br][kb + e];
                c.v = wmma_bf16(a.v, b.v, c.v);
            }
            const int col = n0 + tn + (lane & 15);
            const float bv = bo[col];
            #pragma unroll
            for (int r = 0; r < 8; ++r) {
                const int rr = tm + r + 8 * (lane >> 4);
                out[(row0 + rr) * CZ + col] = c.f[r] + bv;
            }
        }
    }
}

// ===========================================================================
extern "C" void kernel_launch(void* const* d_in, const int* in_sizes, int n_in,
                              void* d_out, int out_size, void* d_ws, size_t ws_size,
                              hipStream_t stream)
{
    (void)in_sizes; (void)n_in; (void)out_size; (void)ws_size;

    const float* z   = (const float*)d_in[0];
    const float* pm  = (const float*)d_in[1];
    const float* lnw = (const float*)d_in[2];
    const float* lnb = (const float*)d_in[3];
    const float* Wq  = (const float*)d_in[4];
    const float* Wk  = (const float*)d_in[5];
    const float* Wv  = (const float*)d_in[6];
    const float* Wb  = (const float*)d_in[7];
    const float* Wg  = (const float*)d_in[8];
    const float* Wo  = (const float*)d_in[9];
    const float* bo  = (const float*)d_in[10];
    float* out = (float*)d_out;

    const size_t SEG = (size_t)65536 * 128 * 2;   // 16 MiB
    char* ws = (char*)d_ws;
    unsigned short* q_ws  = (unsigned short*)(ws + 0 * SEG);
    unsigned short* k_ws  = (unsigned short*)(ws + 1 * SEG);
    unsigned short* v_ws  = (unsigned short*)(ws + 2 * SEG);   // [n][h][d][m]
    unsigned short* g_ws  = (unsigned short*)(ws + 3 * SEG);
    unsigned short* ao_ws = (unsigned short*)(ws + 4 * SEG);
    float*          b_ws  = (float*)(ws + 5 * SEG);

    k1_ln_proj <<<512, 256, 0, stream>>>(z, lnw, lnb, Wq, Wk, Wv, Wb, Wg,
                                         q_ws, k_ws, v_ws, g_ws, b_ws);
    k2_attn    <<<1024, 128, 0, stream>>>(q_ws, k_ws, v_ws, g_ws, b_ws, pm, ao_ws);
    k3_outproj <<<512, 256, 0, stream>>>(ao_ws, Wo, bo, out);
}
